// DPFlashAttention_73907797229785
// MI455X (gfx1250) — compile-verified, matrix operand order checked by reading, not run on hardware
//
#include <hip/hip_runtime.h>
#include <math.h>

typedef __attribute__((ext_vector_type(16))) _Float16 v16h;
typedef __attribute__((ext_vector_type(8)))  _Float16 h8;
typedef __attribute__((ext_vector_type(8)))  float    v8f;
typedef __attribute__((ext_vector_type(4)))  float    f32x4;

static constexpr int Bb = 4, Ss = 2048, Ee = 2048, Hh = 16, Dd = 128;

#define WMMA16x16x32(a, b, c) \
    __builtin_amdgcn_wmma_f32_16x16x32_f16(false, (a), false, (b), (short)0, (c), false, false)

#if defined(__has_builtin)
#if __has_builtin(__builtin_amdgcn_tensor_load_to_lds) && \
    __has_builtin(__builtin_amdgcn_s_wait_tensorcnt)
#define USE_TDM 1
#endif
#endif
#ifndef USE_TDM
#define USE_TDM 0
#endif

__device__ __forceinline__ v8f vzero8() {
    v8f z;
#pragma unroll
    for (int j = 0; j < 8; ++j) z[j] = 0.0f;
    return z;
}

// Per-lane WMMA A/B fragment slice: reads k+[0..7] and k+[16..23] relative to p
// (p already includes the lane's 0/8 k-offset). Two 16B LDS/global loads.
__device__ __forceinline__ v16h load_frag16(const _Float16* p) {
    union { h8 h[2]; v16h v; } u;
    u.h[0] = *(const h8*)p;
    u.h[1] = *(const h8*)(p + 16);
    return u.v;
}

// Butterfly max across the 16-lane half-row using DPP (no LDS traffic).
__device__ __forceinline__ float row_max16(float x) {
    float y;
    y = __builtin_bit_cast(float, __builtin_amdgcn_update_dpp(
            0, __builtin_bit_cast(int, x), 0xB1, 0xF, 0xF, true));   // quad_perm(1,0,3,2)
    x = fmaxf(x, y);
    y = __builtin_bit_cast(float, __builtin_amdgcn_update_dpp(
            0, __builtin_bit_cast(int, x), 0x4E, 0xF, 0xF, true));   // quad_perm(2,3,0,1)
    x = fmaxf(x, y);
    y = __builtin_bit_cast(float, __builtin_amdgcn_update_dpp(
            0, __builtin_bit_cast(int, x), 0x141, 0xF, 0xF, true));  // row_half_mirror
    x = fmaxf(x, y);
    y = __builtin_bit_cast(float, __builtin_amdgcn_update_dpp(
            0, __builtin_bit_cast(int, x), 0x140, 0xF, 0xF, true));  // row_mirror
    x = fmaxf(x, y);
    return x;
}

#if USE_TDM
typedef __attribute__((ext_vector_type(4))) unsigned int u32x4;
typedef __attribute__((ext_vector_type(8))) int          i32x8;
typedef __attribute__((ext_vector_type(4))) int          i32x4;

// Issue a TDM DMA of a 32(keys) x 128(d) f16 tile (row stride Ee elements)
// from global into LDS at byte offset lds_off. Tracked by TENSORcnt.
__device__ __forceinline__ void tdm_load_vtile(const _Float16* gsrc, unsigned lds_off) {
    const unsigned long long ga = (unsigned long long)(size_t)gsrc;
    u32x4 g0 = {};
    g0[0] = 1u;                                              // count=1 user D#
    g0[1] = lds_off;                                         // lds_addr (bytes)
    g0[2] = (unsigned)(ga & 0xFFFFFFFFu);                    // global_addr[31:0]
    g0[3] = (unsigned)((ga >> 32) & 0x01FFFFFFu) | (2u << 30); // addr[56:32] | type=2
    i32x8 g1 = {};
    g1[0] = 1 << 16;                                         // data_size=2B
    g1[1] = (Dd & 0xFFFF) << 16;                             // tensor_dim0=128
    g1[2] = 32 << 16;                                        // tensor_dim1=32
    g1[3] = Dd << 16;                                        // tile_dim0=128
    g1[4] = 32;                                              // tile_dim1=32
    g1[5] = Ee;                                              // tensor_dim0_stride=2048
    i32x4 z4 = {};
#if __clang_major__ >= 23
    i32x8 z8 = {};
    __builtin_amdgcn_tensor_load_to_lds(g0, g1, z4, z4, z8, 0);
#else
    __builtin_amdgcn_tensor_load_to_lds(g0, g1, z4, z4, 0);
#endif
}
#endif

// -------------------------------------------------------------------------
// fp32 -> f16 bulk conversion (one-time; removes all cvt work from GEMM loops)
// -------------------------------------------------------------------------
__global__ __launch_bounds__(256)
void cvt_f32_f16_kernel(const float* __restrict__ src, _Float16* __restrict__ dst) {
    const size_t i = ((size_t)blockIdx.x * 256 + threadIdx.x) * 8;
    f32x4 a = ((const f32x4*)(src + i))[0];
    f32x4 b = ((const f32x4*)(src + i))[1];
    h8 o;
#pragma unroll
    for (int j = 0; j < 4; ++j) { o[j] = (_Float16)a[j]; o[4 + j] = (_Float16)b[j]; }
    *(h8*)(dst + i) = o;
}

// -------------------------------------------------------------------------
// GEMM: Y[M,N] = X[M,K] @ W[N,K]^T + bias[N], X/W already f16.
// Block tile 128x128, 8 waves of 64x32, k-steps of 32. Pure b128 + WMMA loop.
// -------------------------------------------------------------------------
template <typename OutT>
__global__ __launch_bounds__(256)
void gemm_xwT_kernel(const _Float16* __restrict__ X, const _Float16* __restrict__ W,
                     const float* __restrict__ bias, OutT* __restrict__ Y,
                     int M, int N, int K) {
    const int lane = threadIdx.x & 31;
    const int wave = threadIdx.x >> 5;
    const int l16  = lane & 15;
    const int koff = (lane >> 4) << 3;
    const int bm   = blockIdx.y * 128 + (wave >> 2) * 64;
    const int bn   = blockIdx.x * 128 + (wave & 3) * 32;

    const _Float16* xrow[4];
    const _Float16* wrow[2];
#pragma unroll
    for (int mt = 0; mt < 4; ++mt) xrow[mt] = X + (size_t)(bm + mt * 16 + l16) * K + koff;
#pragma unroll
    for (int nt = 0; nt < 2; ++nt) wrow[nt] = W + (size_t)(bn + nt * 16 + l16) * K + koff;

    v8f acc[4][2];
#pragma unroll
    for (int mt = 0; mt < 4; ++mt)
#pragma unroll
        for (int nt = 0; nt < 2; ++nt) acc[mt][nt] = vzero8();

#pragma unroll 1
    for (int k0 = 0; k0 < K; k0 += 32) {
        v16h afrag[4], bfrag[2];
#pragma unroll
        for (int mt = 0; mt < 4; ++mt) afrag[mt] = load_frag16(xrow[mt] + k0);
#pragma unroll
        for (int nt = 0; nt < 2; ++nt) bfrag[nt] = load_frag16(wrow[nt] + k0);
        __builtin_prefetch(xrow[0] + k0 + 32, 0, 1);
        __builtin_prefetch(wrow[0] + k0 + 32, 0, 1);
#pragma unroll
        for (int mt = 0; mt < 4; ++mt)
#pragma unroll
            for (int nt = 0; nt < 2; ++nt)
                acc[mt][nt] = WMMA16x16x32(afrag[mt], bfrag[nt], acc[mt][nt]);
    }

    const int halfrow = (lane >> 4) * 8;
#pragma unroll
    for (int nt = 0; nt < 2; ++nt) {
        const int col = bn + nt * 16 + l16;
        const float bv = bias[col];
#pragma unroll
        for (int mt = 0; mt < 4; ++mt) {
#pragma unroll
            for (int r = 0; r < 8; ++r) {
                const int row = bm + mt * 16 + halfrow + r;
                Y[(size_t)row * N + col] = (OutT)(acc[mt][nt][r] + bv);
            }
        }
    }
}

// -------------------------------------------------------------------------
// Flash attention over f16 Q/K/V ([B,S,H*D] layout), online softmax in f32.
// V tiles are DMA'd by the Tensor Data Mover into a double-buffered LDS region
// one iteration ahead (s_wait_tensorcnt(1) retires the older DMA), then
// transposed in LDS for B-fragment ds_load_b128s. Row-max via DPP, l via
// P@ones WMMA, DP noise fused in the epilogue.
// Grid: (S/128, H, B); 256 threads = 8 waves; each wave owns 16 query rows.
// -------------------------------------------------------------------------
__global__ __launch_bounds__(256)
void flash_attn_kernel(const _Float16* __restrict__ Qw, const _Float16* __restrict__ Kw,
                       const _Float16* __restrict__ Vw, const float* __restrict__ noise,
                       _Float16* __restrict__ Ctx, float scale, float nscale) {
#if USE_TDM
    __shared__ __align__(16) _Float16 VS[2][32 * Dd];   // row-major V tiles, 2x8 KB
#endif
    __shared__ __align__(16) _Float16 VT[Dd * 32];      // V^T tile: [d][key], 8 KB
    __shared__ __align__(16) _Float16 Pb[8][16 * 32];   // per-wave P tile,   8 KB

    const int lane = threadIdx.x & 31;
    const int wave = threadIdx.x >> 5;
    const int l16  = lane & 15;
    const int half = lane >> 4;
    const int koff = half * 8;

    const int b = blockIdx.z, h = blockIdx.y;
    const int q0 = blockIdx.x * 128 + wave * 16;
    const size_t base = ((size_t)b * Ss) * Ee + (size_t)h * Dd;

    v16h qf[4];
    {
        const _Float16* qp = Qw + base + (size_t)(q0 + l16) * Ee;
#pragma unroll
        for (int c = 0; c < 4; ++c) qf[c] = load_frag16(qp + c * 32 + koff);
    }
    v16h ones;
#pragma unroll
    for (int j = 0; j < 16; ++j) ones[j] = (_Float16)1.0f;

    v8f acc[8], acc_l = vzero8();
#pragma unroll
    for (int t = 0; t < 8; ++t) acc[t] = vzero8();
    float mrun[8];
#pragma unroll
    for (int r = 0; r < 8; ++r) mrun[r] = -INFINITY;

#if USE_TDM
    if (wave == 0)   // prologue: DMA first V tile into buffer 0
        tdm_load_vtile(Vw + base, (unsigned)(size_t)&VS[0][0]);
#endif

#pragma unroll 1
    for (int kb = 0; kb < Ss; kb += 32) {
#if USE_TDM
        const int cur = (kb >> 5) & 1;
        if (wave == 0) {
            if (kb + 32 < Ss) {  // prefetch next tile into the other buffer
                tdm_load_vtile(Vw + base + (size_t)(kb + 32) * Ee,
                               (unsigned)(size_t)&VS[cur ^ 1][0]);
                __builtin_amdgcn_s_wait_tensorcnt(1);  // older DMA (tile kb) done
            } else {
                __builtin_amdgcn_s_wait_tensorcnt(0);
            }
        }
        __syncthreads();
        // LDS->LDS transpose: VS[cur] (row-major) -> VT ([d][key]).
        {
            const int t   = threadIdx.x;
            const int key = t >> 3;
            const int d0  = (t & 7) * 16;
            const _Float16* vp = &VS[cur][key * Dd + d0];
            h8 v0 = *(const h8*)vp;
            h8 v1 = *(const h8*)(vp + 8);
#pragma unroll
            for (int j = 0; j < 8; ++j) {
                VT[(d0 + j) * 32 + key]     = v0[j];
                VT[(d0 + 8 + j) * 32 + key] = v1[j];
            }
        }
#else
        // Fallback: transpose-stage V tile straight from global.
        {
            const int t   = threadIdx.x;
            const int key = t >> 3;
            const int d0  = (t & 7) * 16;
            const _Float16* vp = Vw + base + (size_t)(kb + key) * Ee + d0;
            h8 v0 = *(const h8*)vp;
            h8 v1 = *(const h8*)(vp + 8);
#pragma unroll
            for (int j = 0; j < 8; ++j) {
                VT[(d0 + j) * 32 + key]     = v0[j];
                VT[(d0 + 8 + j) * 32 + key] = v1[j];
            }
        }
#endif
        __syncthreads();

        // Scores: Q(16xD) @ K^T(Dx32) -> two 16x16 f32 tiles.
        v8f s0 = vzero8(), s1 = vzero8();
        {
            const _Float16* kp0 = Kw + base + (size_t)(kb + l16) * Ee;
            const _Float16* kp1 = Kw + base + (size_t)(kb + 16 + l16) * Ee;
#pragma unroll
            for (int c = 0; c < 4; ++c) {
                v16h kf0 = load_frag16(kp0 + c * 32 + koff);
                v16h kf1 = load_frag16(kp1 + c * 32 + koff);
                s0 = WMMA16x16x32(qf[c], kf0, s0);
                s1 = WMMA16x16x32(qf[c], kf1, s1);
            }
        }

        // Online softmax: row-max via DPP butterflies; P staged per-wave in LDS.
        float alpha[8];
#pragma unroll
        for (int r = 0; r < 8; ++r) {
            const float x0 = s0[r] * scale, x1 = s1[r] * scale;
            const float rmax = row_max16(fmaxf(x0, x1));
            const float mnew = fmaxf(mrun[r], rmax);
            const float p0 = __expf(x0 - mnew);
            const float p1 = __expf(x1 - mnew);
            alpha[r] = __expf(mrun[r] - mnew);
            mrun[r] = mnew;
            const int row = half * 8 + r;
            Pb[wave][row * 32 + l16]      = (_Float16)p0;
            Pb[wave][row * 32 + 16 + l16] = (_Float16)p1;
        }
#pragma unroll
        for (int r = 0; r < 8; ++r) acc_l[r] *= alpha[r];
#pragma unroll
        for (int t = 0; t < 8; ++t)
#pragma unroll
            for (int r = 0; r < 8; ++r) acc[t][r] *= alpha[r];

        // Wave-private P buffer: LDS ops from one wave are in-order, no barrier.
        const v16h pf = load_frag16(&Pb[wave][l16 * 32 + koff]);
        acc_l = WMMA16x16x32(pf, ones, acc_l);           // l += rowsum(P)
#pragma unroll
        for (int dt = 0; dt < 8; ++dt) {
            const v16h vf = load_frag16(&VT[(dt * 16 + l16) * 32 + koff]);
            acc[dt] = WMMA16x16x32(pf, vf, acc[dt]);
        }
        __syncthreads();
    }

    // Epilogue: normalize, add DP noise, write f16 context.
#pragma unroll
    for (int r = 0; r < 8; ++r) {
        const int row = q0 + half * 8 + r;
        const float inv = 1.0f / acc_l[r];
        const size_t rb = base + (size_t)row * Ee;
#pragma unroll
        for (int dt = 0; dt < 8; ++dt) {
            const int d = dt * 16 + l16;
            const float v = acc[dt][r] * inv + noise[rb + d] * nscale;
            Ctx[rb + d] = (_Float16)v;
        }
    }
}

extern "C" void kernel_launch(void* const* d_in, const int* in_sizes, int n_in,
                              void* d_out, int out_size, void* d_ws, size_t ws_size,
                              hipStream_t stream) {
    (void)in_sizes; (void)n_in; (void)out_size; (void)ws_size;
    const float* query = (const float*)d_in[0];
    const float* key_t = (const float*)d_in[1];
    const float* value = (const float*)d_in[2];
    const float* Wq = (const float*)d_in[3];  const float* bq = (const float*)d_in[4];
    const float* Wk = (const float*)d_in[5];  const float* bk = (const float*)d_in[6];
    const float* Wv = (const float*)d_in[7];  const float* bv = (const float*)d_in[8];
    const float* Wo = (const float*)d_in[9];  const float* bo = (const float*)d_in[10];
    const float* noise = (const float*)d_in[11];
    float* out = (float*)d_out;

    const size_t nAct = (size_t)Bb * Ss * Ee;     // 16.7M elements
    const size_t nWt  = (size_t)Ee * Ee;          // 4.19M elements
    _Float16* qws  = (_Float16*)d_ws;
    _Float16* kws  = qws + nAct;
    _Float16* vws  = kws + nAct;
    _Float16* cws  = vws + nAct;
    _Float16* xtmp = cws + nAct;                  // reused f16 activation staging
    _Float16* wtmp = xtmp + nAct;                 // reused f16 weight staging

    const int M = Bb * Ss;                        // 8192
    dim3 gridG(Ee / 128, M / 128);                // 16 x 64
    const int gAct = (int)(nAct / 2048);          // 8 elems/thread, 256 thr/blk
    const int gWt  = (int)(nWt / 2048);

    cvt_f32_f16_kernel<<<gAct, 256, 0, stream>>>(query, xtmp);
    cvt_f32_f16_kernel<<<gWt, 256, 0, stream>>>(Wq, wtmp);
    gemm_xwT_kernel<_Float16><<<gridG, 256, 0, stream>>>(xtmp, wtmp, bq, qws, M, Ee, Ee);

    cvt_f32_f16_kernel<<<gAct, 256, 0, stream>>>(key_t, xtmp);
    cvt_f32_f16_kernel<<<gWt, 256, 0, stream>>>(Wk, wtmp);
    gemm_xwT_kernel<_Float16><<<gridG, 256, 0, stream>>>(xtmp, wtmp, bk, kws, M, Ee, Ee);

    cvt_f32_f16_kernel<<<gAct, 256, 0, stream>>>(value, xtmp);
    cvt_f32_f16_kernel<<<gWt, 256, 0, stream>>>(Wv, wtmp);
    gemm_xwT_kernel<_Float16><<<gridG, 256, 0, stream>>>(xtmp, wtmp, bv, vws, M, Ee, Ee);

    const float scale  = 1.0f / sqrtf((float)Dd);
    const float nscale = sqrtf(2.0f * logf(1.25f / 1e-5f));  // MAX_GRAD_NORM/EPSILON = 1

    flash_attn_kernel<<<dim3(Ss / 128, Hh, Bb), 256, 0, stream>>>(
        qws, kws, vws, noise, cws, scale, nscale);

    cvt_f32_f16_kernel<<<gWt, 256, 0, stream>>>(Wo, wtmp);
    gemm_xwT_kernel<float><<<gridG, 256, 0, stream>>>(cws, wtmp, bo, out, M, Ee, Ee);
}
